// SelfAttentionConv_2491081032204
// MI455X (gfx1250) — matrix operand order repaired and checked by reading, not compiled, not run on hardware
//
#include <hip/hip_runtime.h>

#define BATCH 2
#define CIN   256
#define OCH   256
#define HH    48
#define WW    48
#define PADK  3
#define KSZ   7
#define KK    49
#define HPD   (HH + 2 * PADK)       // 54
#define WPD   (WW + 2 * PADK)       // 54
#define NPIX  (BATCH * HPD * WPD)   // 5832

typedef __bf16 v16bf __attribute__((ext_vector_type(16)));
typedef __bf16 v8bf  __attribute__((ext_vector_type(8)));
typedef float  v8f   __attribute__((ext_vector_type(8)));

__device__ __forceinline__ __bf16 f2bf(float f)  { return static_cast<__bf16>(f); }
__device__ __forceinline__ float  bf2f(__bf16 b) { return static_cast<float>(b); }

// ---------------------------------------------------------------------------
// Kernel 1: fused dual 1x1-conv GEMM:  qk = Wk * Xpad, v = Wv * Xpad
// One wave computes one 16(M=O) x 16(N=pixel) tile for BOTH outputs,
// sharing the B (x) fragment.  Results stored channel-last bf16.
// ---------------------------------------------------------------------------
__global__ __launch_bounds__(32)
void qkv_gemm_kernel(const float* __restrict__ x,
                     const float* __restrict__ wk,
                     const float* __restrict__ wv,
                     __bf16* __restrict__ qkws,
                     __bf16* __restrict__ vws) {
    const int lane = threadIdx.x;                  // 0..31
    const int mt   = blockIdx.x;                   // O tile 0..15
    const int nt   = blockIdx.y;                   // pixel tile 0..364

    const int  n      = nt * 16 + (lane & 15);
    const bool nvalid = (n < NPIX);
    const int  nc     = nvalid ? n : 0;

    const int b   = nc / (HPD * WPD);
    const int rem = nc % (HPD * WPD);
    const int hp  = rem / WPD, wp = rem % WPD;
    const int hin = hp - PADK, win = wp - PADK;
    const bool inimg = nvalid && ((unsigned)hin < HH) && ((unsigned)win < WW);
    const float* xcol = x + (((size_t)b * CIN) * HH + hin) * WW + win; // +c*HH*WW

    const int kbaseA = (lane >= 16) ? 8 : 0;       // unused; A-frag loads use kbaseB (contiguous)
    const int kbaseB = (lane >= 16) ? 16 : 0;      // B-frag K split
    const int mrow   = mt * 16 + (lane & 15);

    v8f accK = {};
    v8f accV = {};
#pragma unroll
    for (int ks = 0; ks < 8; ++ks) {
        const int c0 = ks * 32;
        v16bf aK, aV, bX = {};
        const float* wkrow = wk + (size_t)mrow * CIN + c0;
        const float* wvrow = wv + (size_t)mrow * CIN + c0;
#pragma unroll
        for (int e = 0; e < 16; ++e) {
            const int kk = e + kbaseB;                                // contiguous c
            aK[e] = f2bf(wkrow[kk]);
            aV[e] = f2bf(wvrow[kk]);
        }
        // single exec toggle around the whole strided channel burst
        if (inimg) {
#pragma unroll
            for (int e = 0; e < 16; ++e) {
                const int kk = e + kbaseB;                            // contiguous c
                bX[e] = f2bf(xcol[(size_t)(c0 + kk) * (HH * WW)]);
            }
        }
        accK = __builtin_amdgcn_wmma_f32_16x16x32_bf16(false, aK, false, bX,
                                                       (short)0, accK, false, false);
        accV = __builtin_amdgcn_wmma_f32_16x16x32_bf16(false, aV, false, bX,
                                                       (short)0, accV, false, false);
    }

    if (nvalid) {
        // D layout: M = r + 8*(lane>=16), N = lane&15 ; 8 rows are contiguous
        // in channel-last storage -> one 16B packed store per output matrix.
        const int moff = mt * 16 + ((lane >= 16) ? 8 : 0);
        v8bf pk, pv;
#pragma unroll
        for (int r = 0; r < 8; ++r) {
            pk[r] = f2bf(accK[r]);
            pv[r] = f2bf(accV[r]);
        }
        *(v8bf*)(qkws + (size_t)n * OCH + moff) = pk;
        *(v8bf*)(vws  + (size_t)n * OCH + moff) = pv;
    }
}

// ---------------------------------------------------------------------------
// Kernel 2: per-pixel attention.  One wave per output pixel.
//   scores = (P + bias)^T P  via 4x4 tiles of wmma bf16 (64x64 padded from 49),
//   softmax rows, column-mean -> wbar[49], out[c] = sum_j wbar[j]*V[c,j].
// ---------------------------------------------------------------------------
__global__ __launch_bounds__(32)
void attn_kernel(const __bf16* __restrict__ qkws,
                 const __bf16* __restrict__ vws,
                 const float* __restrict__ rel_h,
                 const float* __restrict__ rel_w,
                 float* __restrict__ out) {
    __shared__ float S[64][65];
    __shared__ float wbar[64];

    const int lane = threadIdx.x;
    const int p    = blockIdx.x;                   // 0..4607
    const int b    = p / (HH * WW);
    const int rem  = p % (HH * WW);
    const int h    = rem / WW, w = rem % WW;

    const int kbaseA = (lane >= 16) ? 8 : 0;
    const int kbaseB = (lane >= 16) ? 16 : 0;
    const int col    = lane & 15;

    // pixel index in padded field for patch position t
    auto pixidx = [&](int t) -> size_t {
        const int hp = h + t / KSZ, wp = w + t % KSZ;
        return (size_t)((b * HPD + hp) * WPD + wp) * OCH;
    };

    v8f acc[4][4] = {};
    for (int ks = 0; ks < 8; ++ks) {
        const int c0 = ks * 32;
        v16bf afr[4], bfr[4];

        // A fragments: rows = patch index i, K = channel c, plus rel bias
#pragma unroll
        for (int mt = 0; mt < 4; ++mt) {
            v16bf a = {};
            const int i = mt * 16 + col;
            if (i < KK) {
                const int di = i / KSZ, dj = i % KSZ;
                const __bf16* base = qkws + pixidx(i);
#pragma unroll
                for (int e = 0; e < 8; ++e) {
                    const int c = c0 + kbaseB + e;
                    float v = bf2f(base[c]);
                    v += (c < 128) ? rel_h[c * KSZ + di] : rel_w[(c - 128) * KSZ + dj];
                    a[e] = f2bf(v);
                }
#pragma unroll
                for (int e = 0; e < 8; ++e) {
                    const int c = c0 + kbaseB + 8 + e;
                    float v = bf2f(base[c]);
                    v += (c < 128) ? rel_h[c * KSZ + di] : rel_w[(c - 128) * KSZ + dj];
                    a[e + 8] = f2bf(v);
                }
            }
            afr[mt] = a;
        }
        // B fragments: column = patch index j, K = channel c (contiguous bf16)
#pragma unroll
        for (int nt = 0; nt < 4; ++nt) {
            v16bf bfrag = {};
            const int j = nt * 16 + col;
            if (j < KK) {
                const __bf16* bp = qkws + pixidx(j) + c0 + kbaseB;
                bfrag = *(const v16bf*)bp;         // 32B contiguous, aligned
            }
            bfr[nt] = bfrag;
        }
#pragma unroll
        for (int mt = 0; mt < 4; ++mt)
#pragma unroll
            for (int nt = 0; nt < 4; ++nt)
                acc[mt][nt] = __builtin_amdgcn_wmma_f32_16x16x32_bf16(
                    false, afr[mt], false, bfr[nt], (short)0, acc[mt][nt],
                    false, false);
    }

    // spill scores to LDS (D layout: M = r + 8*(lane>=16), N = lane&15)
    const int mhi = (lane >= 16) ? 8 : 0;
#pragma unroll
    for (int mt = 0; mt < 4; ++mt)
#pragma unroll
        for (int nt = 0; nt < 4; ++nt)
#pragma unroll
            for (int r = 0; r < 8; ++r)
                S[mt * 16 + mhi + r][nt * 16 + col] = acc[mt][nt][r];
    __syncthreads();

    // row-wise softmax over valid columns j < 49
    for (int i = lane; i < KK; i += 32) {
        float mx = -1e30f;
        for (int j = 0; j < KK; ++j) mx = fmaxf(mx, S[i][j]);
        float sum = 0.0f;
        for (int j = 0; j < KK; ++j) {
            const float e = __expf(S[i][j] - mx);
            S[i][j] = e;
            sum += e;
        }
        const float inv = 1.0f / sum;
        for (int j = 0; j < KK; ++j) S[i][j] *= inv;
    }
    __syncthreads();

    // column mean over valid rows i < 49
    for (int j = lane; j < 64; j += 32) {
        float s = 0.0f;
        if (j < KK)
            for (int i = 0; i < KK; ++i) s += S[i][j];
        wbar[j] = s * (1.0f / KK);
    }
    __syncthreads();

    // out[c] = sum_j wbar[j] * V[c, patch j]; each lane owns 8 channels
    float o[8] = {};
    const int cbase = lane * 8;
    for (int t = 0; t < KK; ++t) {
        const float wj = wbar[t];
        const __bf16* vp = vws + pixidx(t) + cbase;
        const v8bf vv = *(const v8bf*)vp;          // 16B contiguous
#pragma unroll
        for (int e = 0; e < 8; ++e) o[e] += wj * bf2f(vv[e]);
    }
    float* od = out + ((size_t)b * OCH) * (HH * WW) + h * WW + w;
#pragma unroll
    for (int e = 0; e < 8; ++e) od[(size_t)(cbase + e) * (HH * WW)] = o[e];
}

// ---------------------------------------------------------------------------
extern "C" void kernel_launch(void* const* d_in, const int* in_sizes, int n_in,
                              void* d_out, int out_size, void* d_ws, size_t ws_size,
                              hipStream_t stream) {
    const float* x  = (const float*)d_in[0];
    const float* wk = (const float*)d_in[1];
    const float* wv = (const float*)d_in[2];
    const float* rh = (const float*)d_in[3];
    const float* rw = (const float*)d_in[4];
    float* out = (float*)d_out;

    __bf16* qkws = (__bf16*)d_ws;                         // [NPIX][256] bf16
    __bf16* vws  = qkws + (size_t)NPIX * OCH;             // [NPIX][256] bf16

    dim3 g1(OCH / 16, (NPIX + 15) / 16);                  // 16 x 365 tiles
    qkv_gemm_kernel<<<g1, 32, 0, stream>>>(x, wk, wv, qkws, vws);

    attn_kernel<<<BATCH * HH * WW, 32, 0, stream>>>(qkws, vws, rh, rw, out);
}
